// ASAP_AE_32392643346840
// MI455X (gfx1250) — compile-verified, matrix operand order checked by reading, not run on hardware
//
#include <hip/hip_runtime.h>
#include <hip/hip_bf16.h>
#include <stdint.h>

#define HD   128          // feature width F == H == 128
#define TPB  256

typedef __attribute__((ext_vector_type(16))) __bf16 v16bf;
typedef __attribute__((ext_vector_type(8)))  float  v8f;

// ---------------- small helpers ----------------
__device__ __forceinline__ unsigned short f2bf(float f) {
  unsigned int u = __float_as_uint(f);
  u += 0x7FFFu + ((u >> 16) & 1u);          // round-to-nearest-even
  return (unsigned short)(u >> 16);
}
__device__ __forceinline__ float bf2f(unsigned short b) {
  return __uint_as_float(((unsigned int)b) << 16);
}
__device__ __forceinline__ void atomicMaxF(float* addr, float v) {
  if (v >= 0.0f) atomicMax((int*)addr, __float_as_int(v));
  else           atomicMin((unsigned int*)addr, __float_as_uint(v));
}
__device__ __forceinline__ int srcOf(const int* e, int E, int el) { return el < E ? e[el]     : el - E; }
__device__ __forceinline__ int dstOf(const int* e, int E, int el) { return el < E ? e[E + el] : el - E; }

// ---------------- bf16 WMMA GEMM ----------------
// C[M,N] = opA(A) @ opB(B);  opA = A (TA=0) or A^T-of-stored (TA=1: Aop[m,k]=A[k*lda+m])
//                            opB = B (TB=0) or B^T-of-stored (TB=1: Bop[k,n]=B[n*ldb+k])
// OUTBF: 0 -> f32 out (Cf), 1 -> bf16 out (Cb). M,N multiples of 128; K multiple of 32.
#define BM 128
#define BN 128
#define BK 32
#define LDST 48   // padded LDS row stride (elements); 96B keeps 16B-aligned vector loads

union BFrag { uint4 q[2]; v16bf v; };

template<int TA, int TB, int OUTBF>
__global__ __launch_bounds__(TPB)
void gemm_wmma(const unsigned short* __restrict__ A,
               const unsigned short* __restrict__ B,
               float* __restrict__ Cf, unsigned short* __restrict__ Cb,
               int M, int N, int K, int lda, int ldb, int ldc)
{
  __shared__ unsigned short As[BM * LDST];
  __shared__ unsigned short Bs[BN * LDST];

  const int m0 = blockIdx.y * BM;
  const int n0 = blockIdx.x * BN;
  const int t    = threadIdx.x;
  const int wave = t >> 5;
  const int lane = t & 31;
  const int wm = (wave & 1) * 64;    // wave 2x4 layout: 64 rows x 32 cols each
  const int wn = (wave >> 1) * 32;
  const int fr = lane & 15;          // row (A) / col (B) within fragment
  const int kh = lane >> 4;          // K half selector

  const v8f vzero = {0.f,0.f,0.f,0.f,0.f,0.f,0.f,0.f};
  v8f acc[4][2];
  #pragma unroll
  for (int i = 0; i < 4; ++i)
    #pragma unroll
    for (int j = 0; j < 2; ++j) acc[i][j] = vzero;

  for (int k0 = 0; k0 < K; k0 += BK) {
    { // stage A tile (BM x BK), 16 elems / thread
      int base = t * 16, r = base / BK, c = base % BK;
      if (TA == 0) {
        const unsigned short* g = A + (size_t)(m0 + r) * lda + (k0 + c);
        #pragma unroll
        for (int j = 0; j < 16; ++j) As[r * LDST + c + j] = g[j];
        if (k0 + BK < K)   // speculative prefetch of next K tile (global_prefetch_b8)
          __builtin_prefetch(g + BK, 0, 1);
      } else {
        #pragma unroll
        for (int j = 0; j < 16; ++j)
          As[r * LDST + c + j] = A[(size_t)(k0 + c + j) * lda + (m0 + r)];
        if (k0 + BK < K)
          __builtin_prefetch(A + (size_t)(k0 + BK + c) * lda + (m0 + r), 0, 1);
      }
    }
    { // stage B tile transposed: Bs[n][k] = Bop[k0+k, n0+n]
      int base = t * 16, n = base / BK, c = base % BK;
      if (TB == 1) {
        const unsigned short* g = B + (size_t)(n0 + n) * ldb + (k0 + c);
        #pragma unroll
        for (int j = 0; j < 16; ++j) Bs[n * LDST + c + j] = g[j];
        if (k0 + BK < K)
          __builtin_prefetch(g + BK, 0, 1);
      } else {
        #pragma unroll
        for (int j = 0; j < 16; ++j)
          Bs[n * LDST + c + j] = B[(size_t)(k0 + c + j) * ldb + (n0 + n)];
        if (k0 + BK < K)
          __builtin_prefetch(B + (size_t)(k0 + BK + c) * ldb + (n0 + n), 0, 1);
      }
    }
    __syncthreads();

    BFrag a[4], b[2];
    #pragma unroll
    for (int mf = 0; mf < 4; ++mf) {   // ISA 16-bit A layout: two 8-elem K runs
      const unsigned short* p = &As[(wm + mf * 16 + fr) * LDST + kh * 8];
      a[mf].q[0] = *reinterpret_cast<const uint4*>(p);
      a[mf].q[1] = *reinterpret_cast<const uint4*>(p + 16);
    }
    #pragma unroll
    for (int nf = 0; nf < 2; ++nf) {
      const unsigned short* p = &Bs[(wn + nf * 16 + fr) * LDST + kh * 8];
      b[nf].q[0] = *reinterpret_cast<const uint4*>(p);
      b[nf].q[1] = *reinterpret_cast<const uint4*>(p + 16);
    }
    #pragma unroll
    for (int mf = 0; mf < 4; ++mf)
      #pragma unroll
      for (int nf = 0; nf < 2; ++nf)
        acc[mf][nf] = __builtin_amdgcn_wmma_f32_16x16x32_bf16(
            false, a[mf].v, false, b[nf].v, (short)0, acc[mf][nf], false, false);
    __syncthreads();
  }

  // epilogue: C frag layout — lane: n = lane&15, m = vgpr + (lane>>4)*8
  const int cn = lane & 15;
  const int mo = (lane >> 4) * 8;
  #pragma unroll
  for (int mf = 0; mf < 4; ++mf)
    #pragma unroll
    for (int nf = 0; nf < 2; ++nf) {
      int col = n0 + wn + nf * 16 + cn;
      #pragma unroll
      for (int v = 0; v < 8; ++v) {
        int row = m0 + wm + mf * 16 + mo + v;
        float val = acc[mf][nf][v];
        if (OUTBF) Cb[(size_t)row * ldc + col] = f2bf(val);
        else       Cf[(size_t)row * ldc + col] = val;
      }
    }
}

// ---------------- elementwise / scatter kernels ----------------
__global__ void fill_f32(float* p, float v, long n) {
  long i = (long)blockIdx.x * blockDim.x + threadIdx.x;
  long s = (long)gridDim.x * blockDim.x;
  for (; i < n; i += s) p[i] = v;
}
__global__ void cvt_f32_bf16(const float* in, unsigned short* out, long n) {
  long i = (long)blockIdx.x * blockDim.x + threadIdx.x;
  long s = (long)gridDim.x * blockDim.x;
  for (; i < n; i += s) out[i] = f2bf(in[i]);
}
__global__ void deg_kernel(const int* e, int E, int EL, float* deg) {
  int el = blockIdx.x * blockDim.x + threadIdx.x;
  if (el >= EL) return;
  atomicAdd(&deg[dstOf(e, E, el)], 1.0f);
}
__global__ void dis_kernel(const float* deg, float* dis, int N) {
  int i = blockIdx.x * blockDim.x + threadIdx.x;
  if (i >= N) return;
  dis[i] = rsqrtf(fmaxf(deg[i], 1.0f));
}
__global__ void norm_kernel(const int* e, int E, int EL, const float* dis, float* nv) {
  int el = blockIdx.x * blockDim.x + threadIdx.x;
  if (el >= EL) return;
  nv[el] = dis[srcOf(e, E, el)] * dis[dstOf(e, E, el)];
}
__global__ void scatter_norm(const float* y, const int* e, int E, int EL,
                             const float* nv, float* out) {
  int idx = blockIdx.x * blockDim.x + threadIdx.x;
  if (idx >= EL * (HD / 4)) return;
  int el = idx >> 5, c = (idx & 31) << 2;
  int s = srcOf(e, E, el), d = dstOf(e, E, el);
  float w = nv[el];
  const float4 v = *reinterpret_cast<const float4*>(&y[(size_t)s * HD + c]);
  float* o = &out[(size_t)d * HD + c];
  atomicAdd(o + 0, v.x * w); atomicAdd(o + 1, v.y * w);
  atomicAdd(o + 2, v.z * w); atomicAdd(o + 3, v.w * w);
}
__global__ void bias_act(float* x, const float* b, long n, int act) {
  long i = (long)blockIdx.x * blockDim.x + threadIdx.x;
  long s = (long)gridDim.x * blockDim.x;
  for (; i < n; i += s) {
    float v = x[i] + b[i & (HD - 1)];
    x[i] = act ? tanhf(v) : v;
  }
}
__global__ void seg_max_h(const float* h, const int* e, int E, int EL, float* hmax) {
  int idx = blockIdx.x * blockDim.x + threadIdx.x;
  if (idx >= EL * (HD / 4)) return;
  int el = idx >> 5, c = (idx & 31) << 2;
  int s = srcOf(e, E, el), d = dstOf(e, E, el);
  const float4 v = *reinterpret_cast<const float4*>(&h[(size_t)s * HD + c]);
  float* o = &hmax[(size_t)d * HD + c];
  atomicMaxF(o + 0, v.x); atomicMaxF(o + 1, v.y);
  atomicMaxF(o + 2, v.z); atomicMaxF(o + 3, v.w);
}
__global__ void dots_att(const float* xq, const float* h, const float* att,
                         float* aq, float* ah, int N) {
  int i = blockIdx.x * blockDim.x + threadIdx.x;
  if (i >= N) return;
  float a = 0.f, b = 0.f;
  for (int j = 0; j < HD; ++j) {
    a += att[j]      * xq[(size_t)i * HD + j];
    b += att[HD + j] * h [(size_t)i * HD + j];
  }
  aq[i] = a; ah[i] = b;
}
__global__ void edge_e(const int* e, int E, int EL, const float* aq, const float* ah, float* ev) {
  int el = blockIdx.x * blockDim.x + threadIdx.x;
  if (el >= EL) return;
  float x = aq[dstOf(e, E, el)] + ah[srcOf(e, E, el)];
  ev[el] = x > 0.f ? x : 0.2f * x;               // leaky_relu(0.2)
}
__global__ void seg_max_e(const int* e, int E, int EL, const float* ev, float* emax) {
  int el = blockIdx.x * blockDim.x + threadIdx.x;
  if (el >= EL) return;
  atomicMaxF(&emax[dstOf(e, E, el)], ev[el]);
}
__global__ void edge_exp(const int* e, int E, int EL, const float* ev,
                         const float* emax, float* ee, float* esum) {
  int el = blockIdx.x * blockDim.x + threadIdx.x;
  if (el >= EL) return;
  int d = dstOf(e, E, el);
  float x = __expf(ev[el] - emax[d]);
  ee[el] = x;
  atomicAdd(&esum[d], x);
}
__global__ void edge_alpha(const int* e, int E, int EL, const float* ee,
                           const float* esum, float* alpha) {
  int el = blockIdx.x * blockDim.x + threadIdx.x;
  if (el >= EL) return;
  alpha[el] = ee[el] / esum[dstOf(e, E, el)];
}
__global__ void scatter_cx(const float* h, const int* e, int E, int EL,
                           const float* alpha, float* cx) {
  int idx = blockIdx.x * blockDim.x + threadIdx.x;
  if (idx >= EL * (HD / 4)) return;
  int el = idx >> 5, c = (idx & 31) << 2;
  int s = srcOf(e, E, el), d = dstOf(e, E, el);
  float w = alpha[el];
  const float4 v = *reinterpret_cast<const float4*>(&h[(size_t)s * HD + c]);
  float* o = &cx[(size_t)d * HD + c];
  atomicAdd(o + 0, v.x * w); atomicAdd(o + 1, v.y * w);
  atomicAdd(o + 2, v.z * w); atomicAdd(o + 3, v.w * w);
}
__global__ void dots_le(const float* cx, const float* lw1, const float* lw2,
                        const float* lw3, float* l1, float* l2, float* l3, int N) {
  int i = blockIdx.x * blockDim.x + threadIdx.x;
  if (i >= N) return;
  float a = 0.f, b = 0.f, c = 0.f;
  for (int j = 0; j < HD; ++j) {
    float v = cx[(size_t)i * HD + j];
    a += v * lw1[j]; b += v * lw2[j]; c += v * lw3[j];
  }
  l1[i] = a; l2[i] = b; l3[i] = c;
}
__global__ void scatter_aggr(const int* e, int E, int EL, const float* l2,
                             const float* l3, float* aggr) {
  int el = blockIdx.x * blockDim.x + threadIdx.x;
  if (el >= EL) return;
  int s = srcOf(e, E, el), d = dstOf(e, E, el);
  atomicAdd(&aggr[d], l2[d] - l3[s]);
}
__global__ void fit_kernel(const float* l1, const float* aggr, const float* lb1,
                           float* fit, unsigned long long* keys, int N) {
  int i = blockIdx.x * blockDim.x + threadIdx.x;
  if (i >= N) return;
  float f = 1.0f / (1.0f + __expf(-(l1[i] + lb1[0] + aggr[i])));
  fit[i] = f;
  unsigned int u = __float_as_uint(f);
  unsigned int ord = (u & 0x80000000u) ? ~u : (u | 0x80000000u);   // order-preserving
  keys[i] = ((unsigned long long)ord << 32) | (unsigned int)(0xFFFFFFFFu - (unsigned)i);
}
// exact K-th largest of unique 64-bit keys via 4x16-bit radix select
__global__ void init_ctrl(unsigned long long* ctrl, int Kp) { ctrl[0] = 0ull; ctrl[1] = (unsigned long long)Kp; }
__global__ void hist_kernel(const unsigned long long* keys, int n, unsigned int* hist,
                            const unsigned long long* ctrl, int round) {
  int i = blockIdx.x * blockDim.x + threadIdx.x;
  if (i >= n) return;
  unsigned long long key = keys[i];
  int shift = 48 - 16 * round;
  bool match = true;
  if (round > 0) match = (key >> (shift + 16)) == (ctrl[0] >> (shift + 16));
  if (match) atomicAdd(&hist[(unsigned int)((key >> shift) & 0xFFFFull)], 1u);
}
__global__ void scan_kernel(const unsigned int* hist, unsigned long long* ctrl, int round) {
  if (threadIdx.x || blockIdx.x) return;
  unsigned long long remK = ctrl[1], cum = 0;
  int shift = 48 - 16 * round;
  for (int b = 65535; b >= 0; --b) {
    unsigned long long c = hist[b];
    if (cum + c >= remK) {
      ctrl[0] |= ((unsigned long long)b) << shift;
      ctrl[1]  = remK - cum;
      return;
    }
    cum += c;
  }
}
__global__ void select_kernel(const unsigned long long* keys, const unsigned long long* ctrl,
                              int* sel, int N) {
  int i = blockIdx.x * blockDim.x + threadIdx.x;
  if (i >= N) return;
  sel[i] = keys[i] >= ctrl[0] ? 1 : 0;
}
__global__ void scan_compact(const int* sel, int* inv, int N) {   // 1 block, 1024 thr
  __shared__ int part[1024];
  int t = threadIdx.x;
  int per = (N + 1023) / 1024;
  int s = 0;
  for (int j = 0; j < per; ++j) { int i = t * per + j; if (i < N) s += sel[i]; }
  part[t] = s; __syncthreads();
  for (int off = 1; off < 1024; off <<= 1) {
    int v = (t >= off) ? part[t - off] : 0;
    __syncthreads();
    part[t] += v;
    __syncthreads();
  }
  int run = (t == 0) ? 0 : part[t - 1];
  for (int j = 0; j < per; ++j) {
    int i = t * per + j;
    if (i < N) { inv[i] = sel[i] ? run : -1; run += sel[i]; }
  }
}
__global__ void build_xp(const float* cx, const float* fit, const int* sel,
                         const int* inv, unsigned short* xpbf, int N) {
  int idx = blockIdx.x * blockDim.x + threadIdx.x;
  if (idx >= N * HD) return;
  int i = idx >> 7, c = idx & 127;
  if (!sel[i]) return;
  xpbf[(size_t)inv[i] * HD + c] = f2bf(cx[idx] * fit[i]);
}
__global__ void scatter_S(const int* e, int E, int EL, const float* alpha,
                          const int* sel, const int* inv, float* Sf, int Kp) {
  int el = blockIdx.x * blockDim.x + threadIdx.x;
  if (el >= EL) return;
  int s = srcOf(e, E, el), d = dstOf(e, E, el);
  if (!sel[d]) return;
  atomicAdd(&Sf[(size_t)s * Kp + inv[d]], alpha[el]);
}
// AS[s,:] += S[d,:] per A-edge; bf16 source, skip all-zero 8-wide chunks (S is ~0.4% dense)
__global__ void scatter_AS(const int* e, int E, int EL, const unsigned short* Sbf,
                           float* ASf, int Kp) {
  long idx = (long)blockIdx.x * blockDim.x + threadIdx.x;
  int chunks = Kp >> 3;
  long total = (long)EL * chunks;
  if (idx >= total) return;
  int el = (int)(idx / chunks);
  int c  = (int)(idx % chunks) << 3;
  int s = srcOf(e, E, el), d = dstOf(e, E, el);
  const uint4 v = *reinterpret_cast<const uint4*>(&Sbf[(size_t)d * Kp + c]);
  if ((v.x | v.y | v.z | v.w) == 0u) return;
  const unsigned short* pv = reinterpret_cast<const unsigned short*>(&v);
  float* o = &ASf[(size_t)s * Kp + c];
  #pragma unroll
  for (int j = 0; j < 8; ++j) {
    float f = bf2f(pv[j]);
    if (f != 0.0f) atomicAdd(o + j, f);
  }
}

// ---------------- host-side GCN layer helper ----------------
static void run_gcn(const float* x, const unsigned short* wbf, const float* bias,
                    float* out, float* ybuf, unsigned short* xbf,
                    const int* edges, int E, int N, int EL, const float* nv,
                    int act, hipStream_t stream) {
  long nh = (long)N * HD;
  cvt_f32_bf16<<<4096, TPB, 0, stream>>>(x, xbf, nh);
  gemm_wmma<0, 0, 0><<<dim3(1, N / 128), TPB, 0, stream>>>(
      xbf, wbf, ybuf, nullptr, N, HD, HD, HD, HD, HD);
  hipMemsetAsync(out, 0, nh * sizeof(float), stream);
  int tot = EL * (HD / 4);
  scatter_norm<<<(tot + TPB - 1) / TPB, TPB, 0, stream>>>(ybuf, edges, E, EL, nv, out);
  bias_act<<<4096, TPB, 0, stream>>>(out, bias, nh, act);
}

// ---------------- entry point ----------------
extern "C" void kernel_launch(void* const* d_in, const int* in_sizes, int n_in,
                              void* d_out, int out_size, void* d_ws, size_t ws_size,
                              hipStream_t stream) {
  const float* nodes = (const float*)d_in[0];
  const int*   edges = (const int*)  d_in[1];
  const float* w1 = (const float*)d_in[3];  const float* b1 = (const float*)d_in[4];
  const float* w2 = (const float*)d_in[5];  const float* b2 = (const float*)d_in[6];
  const float* w3 = (const float*)d_in[7];  const float* b3 = (const float*)d_in[8];
  const float* w4 = (const float*)d_in[9];  const float* b4 = (const float*)d_in[10];
  const float* w5 = (const float*)d_in[11]; const float* b5 = (const float*)d_in[12];
  const float* wq = (const float*)d_in[13]; const float* bq = (const float*)d_in[14];
  const float* att = (const float*)d_in[15];
  const float* lw1 = (const float*)d_in[16]; const float* lb1 = (const float*)d_in[17];
  const float* lw2 = (const float*)d_in[18]; const float* lw3 = (const float*)d_in[19];

  const int N  = in_sizes[0] / HD;     // 8192
  const int E  = in_sizes[1] / 2;      // 131072
  const int Kp = N / 2;                // 4096
  const int EL = E + N;                // edges + self loops
  const long NH = (long)N * HD;
  const long NK = (long)N * Kp;

  // bump allocator over d_ws
  char* wsp = (char*)d_ws;
  auto alloc = [&](size_t bytes) -> void* {
    void* r = (void*)wsp;
    wsp += (bytes + 255) & ~(size_t)255;
    return r;
  };
  float* deg   = (float*)alloc(N * 4);
  float* dis   = (float*)alloc(N * 4);
  float* nv    = (float*)alloc((size_t)EL * 4);
  unsigned short* xbf = (unsigned short*)alloc(NH * 2);
  float* ybuf  = (float*)alloc(NH * 4);
  float* hA    = (float*)alloc(NH * 4);
  float* hB    = (float*)alloc(NH * 4);
  unsigned short* wbf1 = (unsigned short*)alloc(HD * HD * 2);
  unsigned short* wbf2 = (unsigned short*)alloc(HD * HD * 2);
  unsigned short* wbf3 = (unsigned short*)alloc(HD * HD * 2);
  unsigned short* wbf4 = (unsigned short*)alloc(HD * HD * 2);
  unsigned short* wbf5 = (unsigned short*)alloc(HD * HD * 2);
  unsigned short* wbfq = (unsigned short*)alloc(HD * HD * 2);
  float* hmax  = (float*)alloc(NH * 4);
  float* xq    = (float*)alloc(NH * 4);
  float* aq    = (float*)alloc(N * 4);
  float* ah    = (float*)alloc(N * 4);
  float* ev    = (float*)alloc((size_t)EL * 4);
  float* ee    = (float*)alloc((size_t)EL * 4);
  float* alpha = (float*)alloc((size_t)EL * 4);
  float* emax  = (float*)alloc(N * 4);
  float* esum  = (float*)alloc(N * 4);
  float* cx    = (float*)alloc(NH * 4);
  float* l1    = (float*)alloc(N * 4);
  float* l2    = (float*)alloc(N * 4);
  float* l3    = (float*)alloc(N * 4);
  float* aggr  = (float*)alloc(N * 4);
  float* fit   = (float*)alloc(N * 4);
  unsigned long long* keys = (unsigned long long*)alloc((size_t)N * 8);
  unsigned int* hist       = (unsigned int*)alloc(65536u * 4);
  unsigned long long* ctrl = (unsigned long long*)alloc(16);
  int* sel  = (int*)alloc(N * 4);
  int* inv  = (int*)alloc(N * 4);
  unsigned short* xpbf = (unsigned short*)alloc((size_t)Kp * HD * 2);
  float* Sf            = (float*)alloc(NK * 4);            // reused later as AS (f32)
  unsigned short* Sbf  = (unsigned short*)alloc(NK * 2);
  unsigned short* ASbf = (unsigned short*)alloc(NK * 2);
  float* Ac            = (float*)alloc((size_t)Kp * Kp * 4);
  unsigned short* Acbf = (unsigned short*)alloc((size_t)Kp * Kp * 2);
  unsigned short* T1bf = (unsigned short*)alloc((size_t)Kp * N * 2);
  float* xout          = (float*)alloc(NH * 4);

  float* x2out  = (float*)d_out;
  float* adjout = (float*)d_out + NH;            // output layout: [x2 | adj_out]

  const int elB = (EL + TPB - 1) / TPB;
  const int nB  = (N + TPB - 1) / TPB;
  const int ehB = (EL * (HD / 4) + TPB - 1) / TPB;

  // ---- degrees + GCN normalization ----
  hipMemsetAsync(deg, 0, N * 4, stream);
  deg_kernel<<<elB, TPB, 0, stream>>>(edges, E, EL, deg);
  dis_kernel<<<nB, TPB, 0, stream>>>(deg, dis, N);
  norm_kernel<<<elB, TPB, 0, stream>>>(edges, E, EL, dis, nv);

  // ---- weights -> bf16 ----
  cvt_f32_bf16<<<64, TPB, 0, stream>>>(w1, wbf1, HD * HD);
  cvt_f32_bf16<<<64, TPB, 0, stream>>>(w2, wbf2, HD * HD);
  cvt_f32_bf16<<<64, TPB, 0, stream>>>(w3, wbf3, HD * HD);
  cvt_f32_bf16<<<64, TPB, 0, stream>>>(w4, wbf4, HD * HD);
  cvt_f32_bf16<<<64, TPB, 0, stream>>>(w5, wbf5, HD * HD);
  cvt_f32_bf16<<<64, TPB, 0, stream>>>(wq, wbfq, HD * HD);

  // ---- encoder: two GCN + tanh layers ----
  run_gcn(nodes, wbf1, b1, hA, ybuf, xbf, edges, E, N, EL, nv, 1, stream);
  run_gcn(hA,    wbf2, b2, hB, ybuf, xbf, edges, E, N, EL, nv, 1, stream);  // hB = h

  // ---- ASAP pooling ----
  fill_f32<<<4096, TPB, 0, stream>>>(hmax, -1e30f, NH);
  seg_max_h<<<ehB, TPB, 0, stream>>>(hB, edges, E, EL, hmax);
  cvt_f32_bf16<<<4096, TPB, 0, stream>>>(hmax, xbf, NH);
  gemm_wmma<0, 0, 0><<<dim3(1, N / 128), TPB, 0, stream>>>(
      xbf, wbfq, xq, nullptr, N, HD, HD, HD, HD, HD);
  bias_act<<<4096, TPB, 0, stream>>>(xq, bq, NH, 0);
  dots_att<<<nB, TPB, 0, stream>>>(xq, hB, att, aq, ah, N);
  edge_e<<<elB, TPB, 0, stream>>>(edges, E, EL, aq, ah, ev);
  fill_f32<<<64, TPB, 0, stream>>>(emax, -1e30f, N);
  seg_max_e<<<elB, TPB, 0, stream>>>(edges, E, EL, ev, emax);
  hipMemsetAsync(esum, 0, N * 4, stream);
  edge_exp<<<elB, TPB, 0, stream>>>(edges, E, EL, ev, emax, ee, esum);
  edge_alpha<<<elB, TPB, 0, stream>>>(edges, E, EL, ee, esum, alpha);
  hipMemsetAsync(cx, 0, NH * 4, stream);
  scatter_cx<<<ehB, TPB, 0, stream>>>(hB, edges, E, EL, alpha, cx);
  dots_le<<<nB, TPB, 0, stream>>>(cx, lw1, lw2, lw3, l1, l2, l3, N);
  hipMemsetAsync(aggr, 0, N * 4, stream);
  scatter_aggr<<<elB, TPB, 0, stream>>>(edges, E, EL, l2, l3, aggr);
  fit_kernel<<<nB, TPB, 0, stream>>>(l1, aggr, lb1, fit, keys, N);

  // ---- exact deterministic top-K membership (radix select on unique 64-bit keys) ----
  init_ctrl<<<1, 1, 0, stream>>>(ctrl, Kp);
  for (int r = 0; r < 4; ++r) {
    hipMemsetAsync(hist, 0, 65536u * 4, stream);
    hist_kernel<<<nB, TPB, 0, stream>>>(keys, N, hist, ctrl, r);
    scan_kernel<<<1, 1, 0, stream>>>(hist, ctrl, r);
  }
  select_kernel<<<nB, TPB, 0, stream>>>(keys, ctrl, sel, N);
  scan_compact<<<1, 1024, 0, stream>>>(sel, inv, N);
  build_xp<<<(int)((NH + TPB - 1) / TPB), TPB, 0, stream>>>(cx, fit, sel, inv, xpbf, N);

  // ---- S (dense f32 scatter -> bf16) ----
  hipMemsetAsync(Sf, 0, NK * 4, stream);
  scatter_S<<<elB, TPB, 0, stream>>>(edges, E, EL, alpha, sel, inv, Sf, Kp);
  cvt_f32_bf16<<<8192, TPB, 0, stream>>>(Sf, Sbf, NK);

  // ---- x_out = S @ xp  (WMMA NN: 8192x128x4096) ----
  gemm_wmma<0, 0, 0><<<dim3(1, N / 128), TPB, 0, stream>>>(
      Sbf, xpbf, xout, nullptr, N, HD, Kp, Kp, HD, HD);

  // ---- AS = A @ S (sparse per-edge scatter, zero-chunk skipping), reuse Sf ----
  hipMemsetAsync(Sf, 0, NK * 4, stream);
  {
    long tot = (long)EL * (Kp >> 3);
    scatter_AS<<<(int)((tot + TPB - 1) / TPB), TPB, 0, stream>>>(edges, E, EL, Sbf, Sf, Kp);
  }
  cvt_f32_bf16<<<8192, TPB, 0, stream>>>(Sf, ASbf, NK);

  // ---- Ac = S^T @ AS (WMMA TN: 4096x4096x8192) ----
  gemm_wmma<1, 0, 0><<<dim3(Kp / 128, Kp / 128), TPB, 0, stream>>>(
      Sbf, ASbf, Ac, nullptr, Kp, Kp, N, Kp, Kp, Kp);
  cvt_f32_bf16<<<8192, TPB, 0, stream>>>(Ac, Acbf, (long)Kp * Kp);

  // ---- T1 = Ac @ S^T (WMMA NT, bf16 out: 4096x8192x4096) ----
  gemm_wmma<0, 1, 1><<<dim3(N / 128, Kp / 128), TPB, 0, stream>>>(
      Acbf, Sbf, nullptr, T1bf, Kp, N, Kp, Kp, Kp, N);

  // ---- adj_out = S @ T1 (WMMA NN, f32 out straight to d_out: 8192x8192x4096) ----
  gemm_wmma<0, 0, 0><<<dim3(N / 128, N / 128), TPB, 0, stream>>>(
      Sbf, T1bf, adjout, nullptr, N, N, Kp, Kp, N, N);

  // ---- decoder: three GCN layers (tanh, tanh, linear) ----
  run_gcn(xout, wbf3, b3, hA, ybuf, xbf, edges, E, N, EL, nv, 1, stream);
  run_gcn(hA,   wbf4, b4, hB, ybuf, xbf, edges, E, N, EL, nv, 1, stream);
  run_gcn(hB,   wbf5, b5, x2out, ybuf, xbf, edges, E, N, EL, nv, 0, stream);
}
